// MoELayer_56298431316475
// MI455X (gfx1250) — matrix operand order repaired and checked by reading, not compiled, not run on hardware
//
#include <hip/hip_runtime.h>
#include <hip/hip_bf16.h>

typedef unsigned short ushort_t;
typedef __attribute__((ext_vector_type(16))) __bf16 v16bf;
typedef __attribute__((ext_vector_type(8)))  float  v8f;

#define T_TOK   32768   // B*S
#define HDIM    512
#define FFNDIM  2048
#define NEXP    64
#define TOPK    2
#define CAPSZ   512
#define NSLOT   (NEXP * CAPSZ)   // 32768
#define NFLAT   (T_TOK * TOPK)   // 65536
#define MT      32               // rows per FFN tile

#if defined(__gfx1250__) && __has_builtin(__builtin_amdgcn_global_load_async_to_lds_b128) && \
    __has_builtin(__builtin_amdgcn_s_wait_asynccnt)
#define USE_ASYNC_LDS 1
typedef int v4i_vec __attribute__((vector_size(16)));
#endif

union BF16x16 { v16bf v; uint4 q[2]; };

static __device__ __forceinline__ ushort_t f2bf(float f) {
  unsigned int u = __float_as_uint(f);
  unsigned int r = u + 0x7FFFu + ((u >> 16) & 1u);   // round-to-nearest-even
  return (ushort_t)(r >> 16);
}

// branch-free tanh-form gelu on the hw transcendental (v_exp_f32)
static __device__ __forceinline__ float gelu_fast(float x) {
  float u  = x * (0.7978845608028654f + 0.035677408136300125f * x * x);
  float t  = __expf(-2.0f * fabsf(u));         // e^{-2|u|}
  float th = (1.0f - t) / (1.0f + t);          // tanh(|u|)
  th = copysignf(th, u);
  return 0.5f * x * (1.0f + th);
}

// ---------------------------------------------------------------- weights -> bf16
__global__ void k_convert_w(const float* __restrict__ w1, const float* __restrict__ w2,
                            ushort_t* __restrict__ w1b, ushort_t* __restrict__ w2b) {
  int i = blockIdx.x * 256 + threadIdx.x;           // 1048576 total
  if (i < FFNDIM * HDIM) {
    w1b[i] = f2bf(w1[i]);
    w2b[i] = f2bf(w2[i]);
  }
}

__global__ void k_zero_buf(uint4* __restrict__ p, int n16) {
  int i = blockIdx.x * 256 + threadIdx.x;
  uint4 z; z.x = 0; z.y = 0; z.z = 0; z.w = 0;
  if (i < n16) p[i] = z;
}

// ---------------------------------------------------------------- router (1 wave / token)
__global__ void __launch_bounds__(256) k_router(const float* __restrict__ x,
                                                const float* __restrict__ rw,
                                                int* __restrict__ eidx,
                                                float* __restrict__ gates) {
  __shared__ float xs[8][HDIM];
  const int wv   = threadIdx.x >> 5;
  const int lane = threadIdx.x & 31;
  const int t    = blockIdx.x * 8 + wv;             // 4096*8 == 32768

  const float* xrow = x + (size_t)t * HDIM;
  #pragma unroll
  for (int i = 0; i < 4; ++i) {
    int c = lane * 16 + i * 4;
    *(float4*)&xs[wv][c] = *(const float4*)&xrow[c];
  }
  __syncthreads();

  const int e0 = lane * 2, e1 = lane * 2 + 1;
  const float* r0 = rw + (size_t)e0 * HDIM;
  const float* r1 = rw + (size_t)e1 * HDIM;
  float d0 = 0.f, d1 = 0.f;
  for (int h = 0; h < HDIM; h += 4) {
    float4 xv = *(const float4*)&xs[wv][h];
    float4 a  = *(const float4*)&r0[h];
    float4 b  = *(const float4*)&r1[h];
    d0 += xv.x*a.x + xv.y*a.y + xv.z*a.z + xv.w*a.w;
    d1 += xv.x*b.x + xv.y*b.y + xv.z*b.z + xv.w*b.w;
  }

  float mx = fmaxf(d0, d1);
  #pragma unroll
  for (int o = 16; o > 0; o >>= 1) mx = fmaxf(mx, __shfl_xor(mx, o, 32));
  float p0 = __expf(d0 - mx), p1 = __expf(d1 - mx);
  float s = p0 + p1;
  #pragma unroll
  for (int o = 16; o > 0; o >>= 1) s += __shfl_xor(s, o, 32);
  p0 /= s; p1 /= s;

  // top-1 (tie -> lower expert id, like argmax-first)
  float bv; int bi;
  if (p0 >= p1) { bv = p0; bi = e0; } else { bv = p1; bi = e1; }
  #pragma unroll
  for (int o = 16; o > 0; o >>= 1) {
    float ov = __shfl_xor(bv, o, 32);
    int   oi = __shfl_xor(bi, o, 32);
    if (ov > bv || (ov == bv && oi < bi)) { bv = ov; bi = oi; }
  }
  // top-2
  float q0 = (e0 == bi) ? -1.0f : p0;
  float q1 = (e1 == bi) ? -1.0f : p1;
  float cv; int ci;
  if (q0 >= q1) { cv = q0; ci = e0; } else { cv = q1; ci = e1; }
  #pragma unroll
  for (int o = 16; o > 0; o >>= 1) {
    float ov = __shfl_xor(cv, o, 32);
    int   oi = __shfl_xor(ci, o, 32);
    if (ov > cv || (ov == cv && oi < ci)) { cv = ov; ci = oi; }
  }

  if (lane == 0) {
    eidx[2*t]   = bi;  gates[2*t]   = bv;
    eidx[2*t+1] = ci;  gates[2*t+1] = cv;
  }
}

// ------------------------------------------- deterministic capacity assignment
// one block per expert: stable running prefix over flat index (matches ref sort)
__global__ void __launch_bounds__(256) k_assign(const int* __restrict__ eidx,
                                                int* __restrict__ slots) {
  const int e    = blockIdx.x;
  const int wv   = threadIdx.x >> 5;
  const int lane = threadIdx.x & 31;
  __shared__ int wsum[8];
  int running = 0;
  for (int base = 0; base < NFLAT; base += 256) {
    int i    = base + threadIdx.x;
    int flag = (eidx[i] == e) ? 1 : 0;
    unsigned mask = (unsigned)__ballot(flag);
    int wpre = __popc(mask & ((1u << lane) - 1u));
    if (lane == 0) wsum[wv] = __popc(mask);
    __syncthreads();
    int wbase = 0, total = 0;
    #pragma unroll
    for (int w = 0; w < 8; ++w) { if (w < wv) wbase += wsum[w]; total += wsum[w]; }
    if (flag) {
      int pos = running + wbase + wpre;
      slots[i] = (pos < CAPSZ) ? (e * CAPSZ + pos) : -1;
    }
    running += total;
    __syncthreads();
  }
}

// ---------------------------------------------------------------- dispatch scatter
__global__ void __launch_bounds__(128) k_scatter(const float* __restrict__ x,
                                                 const int* __restrict__ slots,
                                                 ushort_t* __restrict__ buf) {
  const int i = blockIdx.x;                 // flat assignment index
  const int s = slots[i];
  if (s < 0) return;
  const int tok = i >> 1;
  const float* xr = x + (size_t)tok * HDIM;
  ushort_t* br = buf + (size_t)s * HDIM;
  int c = threadIdx.x * 4;                  // 128 threads * 4 = 512
  float4 v = *(const float4*)&xr[c];
  unsigned lo = (unsigned)f2bf(v.x) | ((unsigned)f2bf(v.y) << 16);
  unsigned hi = (unsigned)f2bf(v.z) | ((unsigned)f2bf(v.w) << 16);
  *(uint2*)&br[c] = make_uint2(lo, hi);
}

// ---------------------------------------------------------------- fused FFN (WMMA)
__global__ void __launch_bounds__(256) k_ffn(const ushort_t* __restrict__ buf,
                                             const ushort_t* __restrict__ w1b,
                                             const float*    __restrict__ b1,
                                             const ushort_t* __restrict__ w2b,
                                             const float*    __restrict__ b2,
                                             float* __restrict__ y) {
  __shared__ ushort_t Atile[MT][HDIM];      //  32 KB
  __shared__ ushort_t Htile[MT][FFNDIM];    // 128 KB  (total 160 KB LDS)

  const int wv    = threadIdx.x >> 5;
  const int lane  = threadIdx.x & 31;
  const int mb    = wv & 1;
  const int g     = wv >> 1;                // 4 column groups of waves
  const int m0    = mb * 16;
  const int l15   = lane & 15;
  const int hi8   = (lane < 16) ? 0 : 8;    // C-fragment row offset per half-wave
  const int koffA = (lane < 16) ? 0 : 16;   // contiguous A k order used here: K 0-15 vs 16-31
  const int baseB = (lane < 16) ? 0 : 16;   // contiguous B k order used here: K 0-15 vs 16-31
  const int row0  = blockIdx.x * MT;

  // stage A tile (rows of the dispatch buffer)
#if defined(USE_ASYNC_LDS)
  {
    const char* gsrc = (const char*)(buf + (size_t)row0 * HDIM);
    char* lbase = (char*)&Atile[0][0];
    for (int i = threadIdx.x; i < (MT * HDIM * 2) / 16; i += 256) {
      __builtin_amdgcn_global_load_async_to_lds_b128(
          (v4i_vec*)(gsrc + (size_t)i * 16),
          (v4i_vec*)(lbase + (size_t)i * 16), 0, 0);
    }
    __builtin_amdgcn_s_wait_asynccnt(0);
  }
#else
  {
    const uint4* src = (const uint4*)(buf + (size_t)row0 * HDIM);
    uint4* dst = (uint4*)&Atile[0][0];
    for (int i = threadIdx.x; i < (MT * HDIM) / 8; i += 256) dst[i] = src[i];
  }
#endif
  __syncthreads();

  // ---- GEMM1: h = gelu(A[32x512] * w1^T + b1) -> Htile (bf16)
  // pairs of 16-col tiles: one A fragment feeds two WMMAs
  for (int jj = 0; jj < 16; ++jj) {
    const int p  = g + jj * 4;              // pair index 0..63
    const int n0 = p * 32 + l15;
    const int n1 = n0 + 16;
    const ushort_t* bp0 = w1b + (size_t)n0 * HDIM + baseB;
    const ushort_t* bp1 = w1b + (size_t)n1 * HDIM + baseB;
    v8f acc0 = {}, acc1 = {};
    #pragma unroll 2
    for (int kb = 0; kb < HDIM / 32; ++kb) {
      const int k0 = kb * 32;
      BF16x16 A, B0, B1;
      A.q[0]  = *(const uint4*)&Atile[m0 + l15][k0 + koffA];
      A.q[1]  = *(const uint4*)&Atile[m0 + l15][k0 + koffA + 8];
      B0.q[0] = *(const uint4*)(bp0 + k0);
      B0.q[1] = *(const uint4*)(bp0 + k0 + 8);
      B1.q[0] = *(const uint4*)(bp1 + k0);
      B1.q[1] = *(const uint4*)(bp1 + k0 + 8);
      __builtin_prefetch(bp0 + k0 + 64, 0, 1);   // global_prefetch_b8
      __builtin_prefetch(bp1 + k0 + 64, 0, 1);
      acc0 = __builtin_amdgcn_wmma_f32_16x16x32_bf16(
               false, A.v, false, B0.v, (short)0, acc0, false, false);
      acc1 = __builtin_amdgcn_wmma_f32_16x16x32_bf16(
               false, A.v, false, B1.v, (short)0, acc1, false, false);
    }
    const float b1n0 = b1[n0];
    const float b1n1 = b1[n1];
    #pragma unroll
    for (int r = 0; r < 8; ++r) {
      Htile[m0 + r + hi8][n0] = f2bf(gelu_fast(acc0[r] + b1n0));
      Htile[m0 + r + hi8][n1] = f2bf(gelu_fast(acc1[r] + b1n1));
    }
  }
  __syncthreads();

  // ---- GEMM2: y = h[32x2048] * w2^T + b2
  for (int jj = 0; jj < 4; ++jj) {
    const int p  = g + jj * 4;              // pair index 0..15
    const int n0 = p * 32 + l15;
    const int n1 = n0 + 16;
    const ushort_t* bp0 = w2b + (size_t)n0 * FFNDIM + baseB;
    const ushort_t* bp1 = w2b + (size_t)n1 * FFNDIM + baseB;
    v8f acc0 = {}, acc1 = {};
    #pragma unroll 2
    for (int kb = 0; kb < FFNDIM / 32; ++kb) {
      const int k0 = kb * 32;
      BF16x16 A, B0, B1;
      A.q[0]  = *(const uint4*)&Htile[m0 + l15][k0 + koffA];
      A.q[1]  = *(const uint4*)&Htile[m0 + l15][k0 + koffA + 8];
      B0.q[0] = *(const uint4*)(bp0 + k0);
      B0.q[1] = *(const uint4*)(bp0 + k0 + 8);
      B1.q[0] = *(const uint4*)(bp1 + k0);
      B1.q[1] = *(const uint4*)(bp1 + k0 + 8);
      __builtin_prefetch(bp0 + k0 + 64, 0, 1);
      __builtin_prefetch(bp1 + k0 + 64, 0, 1);
      acc0 = __builtin_amdgcn_wmma_f32_16x16x32_bf16(
               false, A.v, false, B0.v, (short)0, acc0, false, false);
      acc1 = __builtin_amdgcn_wmma_f32_16x16x32_bf16(
               false, A.v, false, B1.v, (short)0, acc1, false, false);
    }
    const float b2n0 = b2[n0];
    const float b2n1 = b2[n1];
    #pragma unroll
    for (int r = 0; r < 8; ++r) {
      y[(size_t)(row0 + m0 + r + hi8) * HDIM + n0] = acc0[r] + b2n0;
      y[(size_t)(row0 + m0 + r + hi8) * HDIM + n1] = acc1[r] + b2n1;
    }
  }
}

// ---------------------------------------------------------------- gated combine
__global__ void __launch_bounds__(128) k_combine(const float* __restrict__ y,
                                                 const int* __restrict__ slots,
                                                 const float* __restrict__ gates,
                                                 float* __restrict__ out) {
  const int t  = blockIdx.x;
  const int s0 = slots[2*t], s1 = slots[2*t+1];
  const float g0 = gates[2*t], g1 = gates[2*t+1];
  const int c = threadIdx.x * 4;
  float ax = 0.f, ay = 0.f, az = 0.f, aw = 0.f;
  if (s0 >= 0) {
    float4 v = *(const float4*)&y[(size_t)s0 * HDIM + c];
    ax += g0 * v.x; ay += g0 * v.y; az += g0 * v.z; aw += g0 * v.w;
  }
  if (s1 >= 0) {
    float4 v = *(const float4*)&y[(size_t)s1 * HDIM + c];
    ax += g1 * v.x; ay += g1 * v.y; az += g1 * v.z; aw += g1 * v.w;
  }
  float4 o; o.x = ax; o.y = ay; o.z = az; o.w = aw;
  *(float4*)&out[(size_t)t * HDIM + c] = o;
}

// ---------------------------------------------------------------- launcher
extern "C" void kernel_launch(void* const* d_in, const int* in_sizes, int n_in,
                              void* d_out, int out_size, void* d_ws, size_t ws_size,
                              hipStream_t stream) {
  const float* x   = (const float*)d_in[0];   // [T, 512]
  const float* rw  = (const float*)d_in[1];   // [64, 512]
  const float* w1  = (const float*)d_in[2];   // [2048, 512]
  const float* b1  = (const float*)d_in[3];   // [2048]
  const float* w2  = (const float*)d_in[4];   // [512, 2048]
  const float* b2  = (const float*)d_in[5];   // [512]
  float* out = (float*)d_out;

  char* ws = (char*)d_ws;
  ushort_t* w1b  = (ushort_t*)(ws);                                   //  2 MB
  ushort_t* w2b  = (ushort_t*)(ws + (2ull  << 20));                   //  2 MB
  ushort_t* buf  = (ushort_t*)(ws + (4ull  << 20));                   // 32 MB
  float*    yb   = (float*)   (ws + (36ull << 20));                   // 64 MB
  int*      eidx = (int*)     (ws + (100ull << 20));                  // 256 KB
  float*    gts  = (float*)   (ws + (100ull << 20) + (256u << 10));   // 256 KB
  int*      slts = (int*)     (ws + (100ull << 20) + (512u << 10));   // 256 KB

  k_convert_w<<<(FFNDIM * HDIM + 255) / 256, 256, 0, stream>>>(w1, w2, w1b, w2b);
  k_zero_buf<<<((NSLOT * HDIM / 8) + 255) / 256, 256, 0, stream>>>((uint4*)buf,
                                                                   NSLOT * HDIM / 8);
  k_router <<<T_TOK / 8, 256, 0, stream>>>(x, rw, eidx, gts);
  k_assign <<<NEXP, 256, 0, stream>>>(eidx, slts);
  k_scatter<<<NFLAT, 128, 0, stream>>>(x, slts, buf);
  k_ffn    <<<NSLOT / MT, 256, 0, stream>>>(buf, w1b, b1, w2b, b2, yb);
  k_combine<<<T_TOK, 128, 0, stream>>>(yb, slts, gts, out);
}
